// AdaptiveMixing_73589969649841
// MI455X (gfx1250) — matrix operand order repaired and checked by reading, not compiled, not run on hardware
//
#include <hip/hip_runtime.h>
#include <hip/hip_bf16.h>

// ---------------------------------------------------------------------------
// AdaptiveMixing for MI455X (gfx1250): f16 WMMA pipeline, f16 intermediates,
// async global->LDS double-buffered streaming in the final GEMM.
// ---------------------------------------------------------------------------

typedef __attribute__((ext_vector_type(16))) _Float16 v16h;
typedef __attribute__((ext_vector_type(8)))  _Float16 v8h;
typedef __attribute__((ext_vector_type(8)))  float    v8f;

#define WMMA16(a, b, c) __builtin_amdgcn_wmma_f32_16x16x32_f16( \
    false, (a), false, (b), (short)0, (c), false, false)

// Problem constants
constexpr int NQ    = 4 * 900;        // 3600 queries
constexpr int QD    = 256;
constexpr int Gg    = 4;
constexpr int Pp    = 32;
constexpr int Cc    = 64;
constexpr int PO    = 128;
constexpr int CO    = 64;
constexpr int TOTAL = Cc * CO + Pp * PO;  // 8192
constexpr int GP    = Gg * TOTAL;         // 32768
constexpr float EPSI = 1e-5f;

// ---------------------------------------------------------------------------
// WMMA fragment loaders (wave32, v_wmma_f32_16x16x32_f16 layouts)
// A 16x32 f16: lane holds row = lane%16; halves 0..7 -> K = c0..c0+7,
// halves 8..15 -> K = c0+16..c0+23, c0 = (lane>=16) ? 8 : 0.
// B 32x16 f16: lane holds col = lane%16; same K mapping as A.
// C/D f32:     reg r -> (M = r + 8*(lane>=16), N = lane%16).
// ---------------------------------------------------------------------------
__device__ __forceinline__ v16h load_a16(const _Float16* p, int ld, int lane) {
  const _Float16* r = p + (lane & 15) * ld + ((lane >> 4) << 3);
  v8h lo = *(const v8h*)(r);
  v8h hi = *(const v8h*)(r + 16);
  v16h a;
#pragma unroll
  for (int i = 0; i < 8; ++i) { a[i] = lo[i]; a[i + 8] = hi[i]; }
  return a;
}

__device__ __forceinline__ v16h load_b16(const _Float16* p, long ld, int lane) {
  const _Float16* r = p + (long)(lane & 15) * ld + ((lane >> 4) << 3);
  v8h lo = *(const v8h*)(r);
  v8h hi = *(const v8h*)(r + 16);
  v16h b;
#pragma unroll
  for (int i = 0; i < 8; ++i) { b[i] = lo[i]; b[i + 8] = hi[i]; }
  return b;
}

// Async global -> LDS copy of 16 bytes per lane (gfx1250, tracked by ASYNCcnt).
// lds_off is the LDS byte offset (low 32 bits of the generic shared pointer,
// per the flat->LDS aperture mapping).
__device__ __forceinline__ void async_ld_b128(unsigned lds_off, const void* g) {
  asm volatile("global_load_async_to_lds_b128 %0, %1, off"
               :: "v"(lds_off), "v"(g) : "memory");
}

// Block-wide sum + sumsq reduction over 256 threads.
__device__ __forceinline__ void block_reduce2(float& s, float& q,
                                              float* red, float* red2) {
  const int tid = threadIdx.x;
  red[tid] = s; red2[tid] = q;
  __syncthreads();
#pragma unroll
  for (int off = 128; off > 0; off >>= 1) {
    if (tid < off) { red[tid] += red[tid + off]; red2[tid] += red2[tid + off]; }
    __syncthreads();
  }
  s = red[0]; q = red2[0];
  __syncthreads();
}

// ---------------------------------------------------------------------------
// Kernel 0a: generic f32 -> f16 pack
// ---------------------------------------------------------------------------
__global__ void k_cvt(const float* __restrict__ src, _Float16* __restrict__ dst,
                      int n) {
  int i = blockIdx.x * blockDim.x + threadIdx.x;
  if (i < n) dst[i] = (_Float16)src[i];
}

// Kernel 0b: q16 = f16(query + query_pos)
__global__ void k_addcvt(const float* __restrict__ a, const float* __restrict__ b,
                         _Float16* __restrict__ dst, int n) {
  int i = blockIdx.x * blockDim.x + threadIdx.x;
  if (i < n) dst[i] = (_Float16)(a[i] + b[i]);
}

// ---------------------------------------------------------------------------
// Kernel 1: params16[n][j] = f16( q16[n] . Wg16[j] + bg[j] )
// Block: 256 thr (8 waves). Tile: M=16 queries x N=256 params, K=256.
// Wave w covers N = [w*32, w*32+32) -> 2 WMMA column tiles.
// Wg16 rows are already the WMMA B-fragment layout (L2-resident).
// ---------------------------------------------------------------------------
__global__ void k_gemm1(const _Float16* __restrict__ q16,
                        const _Float16* __restrict__ Wg16,
                        const float* __restrict__ bg,
                        _Float16* __restrict__ params16) {
  __shared__ __align__(16) _Float16 qs[16 * 256];  // 8 KB A tile
  const int tid  = threadIdx.x;
  const int lane = tid & 31;
  const int w    = tid >> 5;
  const int m0   = blockIdx.y * 16;         // query tile
  const int n0   = blockIdx.x * 256;        // param tile

  // Stage A tile: 16 rows x 256 halves; each thread copies 32 bytes.
  {
    const int row = tid >> 4, ko = (tid & 15) * 16;
    const v8h* s = (const v8h*)(q16 + (size_t)(m0 + row) * QD + ko);
    v8h* d = (v8h*)&qs[row * 256 + ko];
    d[0] = s[0]; d[1] = s[1];
  }
  __syncthreads();

  const int ntb0 = n0 + w * 32, ntb1 = ntb0 + 16;
  v8f acc0 = {}, acc1 = {};
#pragma unroll
  for (int k0 = 0; k0 < 256; k0 += 32) {
    v16h a  = load_a16(qs + k0, 256, lane);
    v16h b0 = load_b16(Wg16 + (size_t)ntb0 * QD + k0, QD, lane);
    v16h b1 = load_b16(Wg16 + (size_t)ntb1 * QD + k0, QD, lane);
    acc0 = WMMA16(a, b0, acc0);
    acc1 = WMMA16(a, b1, acc1);
  }

  const int p0 = ntb0 + (lane & 15);
  const int p1 = ntb1 + (lane & 15);
  const float bb0 = bg[p0], bb1 = bg[p1];
#pragma unroll
  for (int r = 0; r < 8; ++r) {
    const int M = r + ((lane >> 4) << 3);
    const size_t ro = (size_t)(m0 + M) * GP;
    params16[ro + p0] = (_Float16)(acc0[r] + bb0);
    params16[ro + p1] = (_Float16)(acc1[r] + bb1);
  }
}

// ---------------------------------------------------------------------------
// Kernel 2: per-(n,g) adaptive mixing, fully fused:
//   T1 = x(32x64) @ M(64x64); LN+ReLU; T2 = S(128x32) @ T1; LN+ReLU -> mixed16
// Block: 256 thr (8 waves) per (n,g). 14400 blocks.
// ---------------------------------------------------------------------------
__global__ void k_mix(const float* __restrict__ x,
                      const _Float16* __restrict__ params16,
                      _Float16* __restrict__ mixed16) {
  __shared__ __align__(16) _Float16 xs[Pp * Cc];       // 4 KB  x (p,c) f16
  __shared__ __align__(16) _Float16 mt[CO * Cc];       // 8 KB  M^T (d,c)
  __shared__ __align__(16) _Float16 t1t[CO * Pp];      // 4 KB  T1^T (d,p)
  __shared__ float red[256], red2[256];

  const int tid  = threadIdx.x;
  const int lane = tid & 31;
  const int w    = tid >> 5;
  const int g    = blockIdx.x;
  const int n    = blockIdx.y;
  const size_t pb = (size_t)n * GP + (size_t)g * TOTAL;  // params row base

  // Stage x -> f16 and M -> transposed f16.
  {
    const size_t xb = ((size_t)n * Gg + g) * (Pp * Cc);
#pragma unroll
    for (int i = tid; i < Pp * Cc; i += 256) xs[i] = (_Float16)x[xb + i];
#pragma unroll
    for (int i = tid; i < Cc * CO; i += 256)
      mt[(i & 63) * Cc + (i >> 6)] = params16[pb + i];   // mt[d][c] = M[c][d]
  }
  __syncthreads();

  // ---- Stage 1: T1 = x @ M. 8 tiles (2 M-rows x 4 N-cols), 1 tile/wave.
  const int m1  = w & 1;       // p-tile
  const int nt1 = w >> 1;      // d-tile
  v8f acc = {};
#pragma unroll
  for (int k0 = 0; k0 < Cc; k0 += 32) {
    v16h a = load_a16(xs + m1 * 16 * Cc + k0, Cc, lane);
    v16h b = load_b16(mt + nt1 * 16 * Cc + k0, Cc, lane);
    acc = WMMA16(a, b, acc);
  }

  // LN over all 2048 elements of T1 (from accumulators), ReLU, -> T1^T (f16).
  float s = 0.f, q = 0.f;
#pragma unroll
  for (int r = 0; r < 8; ++r) { s += acc[r]; q += acc[r] * acc[r]; }
  block_reduce2(s, q, red, red2);
  {
    const float mu = s * (1.f / 2048.f);
    const float rs = rsqrtf(q * (1.f / 2048.f) - mu * mu + EPSI);
    const int d = nt1 * 16 + (lane & 15);
#pragma unroll
    for (int r = 0; r < 8; ++r) {
      const int p = m1 * 16 + r + ((lane >> 4) << 3);
      t1t[d * Pp + p] = (_Float16)fmaxf((acc[r] - mu) * rs, 0.f);
    }
  }
  __syncthreads();

  // ---- Stage 2: T2 = S @ T1. Wave w: o-rows [16w,16w+16), 4 d-tiles.
  // S (PO x P) is row-major in params16 -> direct global A-fragment loads.
  const _Float16* Sp = params16 + pb + (Cc * CO) + (size_t)(w * 16) * Pp;
  v8f a2[4];
#pragma unroll
  for (int t = 0; t < 4; ++t) a2[t] = (v8f){};
  {
    v16h af = load_a16(Sp, Pp, lane);
#pragma unroll
    for (int t = 0; t < 4; ++t) {
      v16h b = load_b16(t1t + t * 16 * Pp, Pp, lane);
      a2[t] = WMMA16(af, b, a2[t]);
    }
  }

  // LN over all 8192 elements of T2, ReLU, store f16 (g*8192 + o*64 + d).
  s = 0.f; q = 0.f;
#pragma unroll
  for (int t = 0; t < 4; ++t)
#pragma unroll
    for (int r = 0; r < 8; ++r) { s += a2[t][r]; q += a2[t][r] * a2[t][r]; }
  block_reduce2(s, q, red, red2);
  {
    const float mu = s * (1.f / 8192.f);
    const float rs = rsqrtf(q * (1.f / 8192.f) - mu * mu + EPSI);
    _Float16* ob = mixed16 + (size_t)n * GP + (size_t)g * TOTAL;
#pragma unroll
    for (int t = 0; t < 4; ++t) {
      const int d = t * 16 + (lane & 15);
#pragma unroll
      for (int r = 0; r < 8; ++r) {
        const int o = w * 16 + r + ((lane >> 4) << 3);
        ob[o * CO + d] = (_Float16)fmaxf((a2[t][r] - mu) * rs, 0.f);
      }
    }
  }
}

// ---------------------------------------------------------------------------
// Kernel 3: out[n][j] = query[n][j] + bo[j] + mixed16[n] . Wo16[j]
// Block: 256 thr, tile M=16 queries x N=256 (all of QD), K=32768 streamed
// through LDS with DOUBLE-BUFFERED async global->LDS copies (ASYNCcnt).
// mixed16 (236 MB) is read exactly once from HBM; Wo16 is L2-resident.
// Pipeline per chunk i: issue async copies for i+1 into buf[(i+1)&1],
// s_wait_asynccnt 2 (chunk i's 2 wave-issues complete, in-order), barrier,
// WMMA on buf[i&1], barrier (protects buf reuse at i+2's issue).
// ---------------------------------------------------------------------------
__global__ void k_gemm2(const _Float16* __restrict__ mixed16,
                        const _Float16* __restrict__ Wo16,
                        const float* __restrict__ bo,
                        const float* __restrict__ query,
                        float* __restrict__ out) {
  __shared__ __align__(16) _Float16 as[2][16 * 256];  // 2 x 8 KB A chunks
  const int tid  = threadIdx.x;
  const int lane = tid & 31;
  const int w    = tid >> 5;
  const int m0   = blockIdx.x * 16;
  const int j0   = w * 32, j1 = j0 + 16;     // this wave's output columns

  // Chunk = 16 rows x 256 halves = 512 x 16B segments; 2 segments per thread.
  // seg -> row = seg>>5, half-offset = (seg&31)*8.
  const int row0 = tid >> 5,        ko0 = (tid & 31) * 8;
  const int row1 = (tid + 256) >> 5, ko1 = (tid & 31) * 8;  // rows 8..15
  const _Float16* g0 = mixed16 + (size_t)(m0 + row0) * GP + ko0;
  const _Float16* g1 = mixed16 + (size_t)(m0 + row1) * GP + ko1;

  v8f acc0 = {}, acc1 = {};
  const int NCH = GP / 256;  // 128 chunks

  // Prime the pipeline: chunk 0 -> buffer 0.
  async_ld_b128((unsigned)(uintptr_t)&as[0][row0 * 256 + ko0], g0);
  async_ld_b128((unsigned)(uintptr_t)&as[0][row1 * 256 + ko1], g1);

  for (int i = 0; i < NCH; ++i) {
    const int kc = i * 256;
    if (i + 1 < NCH) {
      const int nb = (i + 1) & 1, nk = kc + 256;
      async_ld_b128((unsigned)(uintptr_t)&as[nb][row0 * 256 + ko0], g0 + nk);
      async_ld_b128((unsigned)(uintptr_t)&as[nb][row1 * 256 + ko1], g1 + nk);
      asm volatile("s_wait_asynccnt 2" ::: "memory");  // chunk i landed
    } else {
      asm volatile("s_wait_asynccnt 0" ::: "memory");
    }
    __syncthreads();

    const _Float16* ab = as[i & 1];
#pragma unroll
    for (int kk = 0; kk < 256; kk += 32) {
      v16h a  = load_a16(ab + kk, 256, lane);
      v16h b0 = load_b16(Wo16 + (size_t)j0 * GP + kc + kk, GP, lane);
      v16h b1 = load_b16(Wo16 + (size_t)j1 * GP + kc + kk, GP, lane);
      acc0 = WMMA16(a, b0, acc0);
      acc1 = WMMA16(a, b1, acc1);
    }
    __syncthreads();  // all waves done with buf[i&1] before it is refilled
  }

  const int c0 = j0 + (lane & 15), c1 = j1 + (lane & 15);
  const float b0 = bo[c0], b1 = bo[c1];
#pragma unroll
  for (int r = 0; r < 8; ++r) {
    const int row = m0 + r + ((lane >> 4) << 3);
    out[row * QD + c0] = query[row * QD + c0] + b0 + acc0[r];
    out[row * QD + c1] = query[row * QD + c1] + b1 + acc1[r];
  }
}

// ---------------------------------------------------------------------------
// Host launch
// ---------------------------------------------------------------------------
extern "C" void kernel_launch(void* const* d_in, const int* in_sizes, int n_in,
                              void* d_out, int out_size, void* d_ws, size_t ws_size,
                              hipStream_t stream) {
  const float* x     = (const float*)d_in[0];
  const float* query = (const float*)d_in[1];
  const float* qpos  = (const float*)d_in[2];
  const float* Wg    = (const float*)d_in[3];
  const float* bg    = (const float*)d_in[4];
  const float* Wo    = (const float*)d_in[5];
  const float* bo    = (const float*)d_in[6];
  float* out = (float*)d_out;

  // Workspace layout (f16 intermediates; all offsets 16B-aligned):
  //   q16      : NQ*QD            =   921,600 elems
  //   Wg16     : GP*QD            = 8,388,608
  //   Wo16     : QD*GP            = 8,388,608
  //   params16 : NQ*GP            = 117,964,800
  //   mixed16  : NQ*GP            = 117,964,800
  char* ws = (char*)d_ws;
  _Float16* q16      = (_Float16*)(ws);
  _Float16* Wg16     = (_Float16*)(ws + 1843200);
  _Float16* Wo16     = (_Float16*)(ws + 18620416);
  _Float16* params16 = (_Float16*)(ws + 35397632);
  _Float16* mixed16  = (_Float16*)(ws + 271327232);

  const int nq  = NQ * QD;       // 921600
  const int nwg = GP * QD;       // 8388608

  k_addcvt<<<(nq + 255) / 256, 256, 0, stream>>>(query, qpos, q16, nq);
  k_cvt<<<(nwg + 255) / 256, 256, 0, stream>>>(Wg, Wg16, nwg);
  k_cvt<<<(nwg + 255) / 256, 256, 0, stream>>>(Wo, Wo16, nwg);

  k_gemm1<<<dim3(GP / 256, NQ / 16), 256, 0, stream>>>(q16, Wg16, bg, params16);
  k_mix<<<dim3(Gg, NQ), 256, 0, stream>>>(x, params16, mixed16);
  k_gemm2<<<NQ / 16, 256, 0, stream>>>(mixed16, Wo16, bo, query, out);
}
